// DyadicQALoRA_15865609192025
// MI455X (gfx1250) — 8-bit matrix operand order repaired and checked by reading, compiled, not run on hardware
//
#include <hip/hip_runtime.h>
#include <stdint.h>

typedef __attribute__((ext_vector_type(8))) int v8i;
typedef __attribute__((ext_vector_type(4))) int v4i;

#define D_IN   4096
#define D_OUT  4096
#define M_TOK  8192
#define QMAXF  127.0f
#define EPSQ   1e-6f

#define LDS_STRIDE 80

#define OFF_SCALES 0u
#define OFF_PART   256u
#define OFF_XSCALE 8192u
#define OFF_T      40960u
#define OFF_AQT    565248u
#define OFF_BQT    630784u
#define OFF_XQ     696320u
#define OFF_WQ     34250752u

#if __has_builtin(__builtin_amdgcn_global_load_async_to_lds_b128)
#define HAVE_ASYNC_LDS 1
#else
#define HAVE_ASYNC_LDS 0
#endif

#if HAVE_ASYNC_LDS
typedef __attribute__((address_space(1))) v4i gv4i;
typedef __attribute__((address_space(3))) v4i lv4i;

__device__ __forceinline__ void async_copy16(const signed char* g, signed char* l) {
  __builtin_amdgcn_global_load_async_to_lds_b128(
      (gv4i*)(uintptr_t)g,
      (lv4i*)(uint32_t)(uintptr_t)l,
      0, 0);
}
#endif

__device__ __forceinline__ void wait_async0() {
#if __has_builtin(__builtin_amdgcn_s_wait_asynccnt)
  __builtin_amdgcn_s_wait_asynccnt(0);
#else
  asm volatile("s_wait_asynccnt 0x0" ::: "memory");
#endif
}

__device__ __forceinline__ v8i load_a_frag(const signed char* base, int row0, int k0, int lane) {
  int m  = row0 + (lane & 15);
  int kb = k0 + ((lane >> 4) << 4);
  const signed char* p = base + (size_t)m * D_IN + kb;
  union { int4 q[2]; v8i v; } u;
  u.q[0] = *(const int4*)(p);
  u.q[1] = *(const int4*)(p + 32);
  return u.v;
}

__device__ __forceinline__ v8i load_b_frag(const signed char* base, int col0, int k0, int lane, int ldb) {
  int n  = col0 + (lane & 15);
  int kb = k0 + ((lane >> 4) << 4);
  const signed char* p = base + (size_t)n * ldb + kb;
  union { int4 q[2]; v8i v; } u;
  u.q[0] = *(const int4*)(p);
  u.q[1] = *(const int4*)(p + 32);
  return u.v;
}

__device__ __forceinline__ v8i lds_a_frag(const signed char* tile, int row0, int lane) {
  const signed char* p = tile + (row0 + (lane & 15)) * LDS_STRIDE + ((lane >> 4) << 4);
  union { int4 q[2]; v8i v; } u;
  u.q[0] = *(const int4*)(p);
  u.q[1] = *(const int4*)(p + 32);
  return u.v;
}

__device__ __forceinline__ v8i lds_b_frag(const signed char* tile, int col0, int lane) {
  const signed char* p = tile + (col0 + (lane & 15)) * LDS_STRIDE + ((lane >> 4) << 4);
  union { int4 q[2]; v8i v; } u;
  u.q[0] = *(const int4*)(p);
  u.q[1] = *(const int4*)(p + 32);
  return u.v;
}

__global__ __launch_bounds__(256) void k_quant_act(const float* __restrict__ x,
                                                   signed char* __restrict__ xq,
                                                   float* __restrict__ xscale) {
  __shared__ float red[256];
  const int m = blockIdx.x;
  const int tid = threadIdx.x;
  const float* rp = x + (size_t)m * D_IN + (size_t)tid * 16;
  float vals[16];
#pragma unroll
  for (int i = 0; i < 4; ++i) {
    float4 v = ((const float4*)rp)[i];
    vals[4*i+0] = v.x; vals[4*i+1] = v.y; vals[4*i+2] = v.z; vals[4*i+3] = v.w;
  }
  float mx = 0.f;
#pragma unroll
  for (int i = 0; i < 16; ++i) mx = fmaxf(mx, fabsf(vals[i]));
  red[tid] = mx;
  __syncthreads();
  for (int s = 128; s > 0; s >>= 1) {
    if (tid < s) red[tid] = fmaxf(red[tid], red[tid + s]);
    __syncthreads();
  }
  float s0 = fmaxf(red[0], EPSQ);
  float inv = QMAXF / s0;
  if (tid == 0) xscale[m] = s0 / QMAXF;
  union { signed char c[16]; int4 q; } u;
#pragma unroll
  for (int i = 0; i < 16; ++i) {
    float r = fminf(fmaxf(rintf(vals[i] * inv), -QMAXF), QMAXF);
    u.c[i] = (signed char)(int)r;
  }
  ((int4*)(xq + (size_t)m * D_IN))[tid] = u.q;
}

__global__ __launch_bounds__(256) void k_wabs(const float* __restrict__ w, float* __restrict__ part) {
  __shared__ float red[256];
  const int tid = threadIdx.x;
  const float4* p = (const float4*)(w + (size_t)blockIdx.x * 16384);
  float s = 0.f;
#pragma unroll
  for (int i = 0; i < 16; ++i) {
    float4 v = p[tid + 256 * i];
    s += fabsf(v.x) + fabsf(v.y) + fabsf(v.z) + fabsf(v.w);
  }
  red[tid] = s;
  __syncthreads();
  for (int st = 128; st > 0; st >>= 1) {
    if (tid < st) red[tid] += red[tid + st];
    __syncthreads();
  }
  if (tid == 0) part[blockIdx.x] = red[0];
}

__global__ __launch_bounds__(256) void k_finalize(const float* __restrict__ part,
                                                  const float* __restrict__ lA,
                                                  const float* __restrict__ lB,
                                                  float* __restrict__ scales) {
  __shared__ float red[256];
  const int tid = threadIdx.x;
  float s = part[tid] + part[tid + 256] + part[tid + 512] + part[tid + 768];
  red[tid] = s;
  __syncthreads();
  for (int st = 128; st > 0; st >>= 1) {
    if (tid < st) red[tid] += red[tid + st];
    __syncthreads();
  }
  float wsum = red[0];
  __syncthreads();
  const float4* pa = (const float4*)lA;
  float mA = 0.f;
#pragma unroll 4
  for (int i = 0; i < 64; ++i) {
    float4 v = pa[tid + 256 * i];
    mA = fmaxf(mA, fmaxf(fmaxf(fabsf(v.x), fabsf(v.y)), fmaxf(fabsf(v.z), fabsf(v.w))));
  }
  red[tid] = mA;
  __syncthreads();
  for (int st = 128; st > 0; st >>= 1) {
    if (tid < st) red[tid] = fmaxf(red[tid], red[tid + st]);
    __syncthreads();
  }
  float amax = fmaxf(red[0], EPSQ);
  __syncthreads();
  const float4* pb = (const float4*)lB;
  float mB = 0.f;
#pragma unroll 4
  for (int i = 0; i < 64; ++i) {
    float4 v = pb[tid + 256 * i];
    mB = fmaxf(mB, fmaxf(fmaxf(fabsf(v.x), fabsf(v.y)), fmaxf(fabsf(v.z), fabsf(v.w))));
  }
  red[tid] = mB;
  __syncthreads();
  for (int st = 128; st > 0; st >>= 1) {
    if (tid < st) red[tid] = fmaxf(red[tid], red[tid + st]);
    __syncthreads();
  }
  float bmax = fmaxf(red[0], EPSQ);
  if (tid == 0) {
    float w_scale = fmaxf(wsum * (1.0f / 16777216.0f), EPSQ);
    scales[0] = w_scale;
    scales[1] = 1.0f / w_scale;
    scales[2] = QMAXF / amax;
    scales[3] = QMAXF / bmax;
    scales[4] = (amax / QMAXF) * (bmax / QMAXF) * 2.0f;
  }
}

__global__ __launch_bounds__(256) void k_quant_w(const float* __restrict__ w,
                                                 signed char* __restrict__ wq,
                                                 const float* __restrict__ scales) {
  const float inv = scales[1];
  const size_t e4 = (size_t)blockIdx.x * 256 + threadIdx.x;
  float4 v = ((const float4*)w)[e4];
  union { signed char c[4]; int i; } u;
  float r;
  r = fminf(fmaxf(rintf(v.x * inv), -1.f), 1.f); u.c[0] = (signed char)(int)r;
  r = fminf(fmaxf(rintf(v.y * inv), -1.f), 1.f); u.c[1] = (signed char)(int)r;
  r = fminf(fmaxf(rintf(v.z * inv), -1.f), 1.f); u.c[2] = (signed char)(int)r;
  r = fminf(fmaxf(rintf(v.w * inv), -1.f), 1.f); u.c[3] = (signed char)(int)r;
  ((int*)wq)[e4] = u.i;
}

__global__ __launch_bounds__(256) void k_quant_lA(const float* __restrict__ lA,
                                                  signed char* __restrict__ aqT,
                                                  const float* __restrict__ scales) {
  const float inv = scales[2];
  const int e = blockIdx.x * 256 + threadIdx.x;
  const int r = e >> 12, k = e & 4095;
  float q = fminf(fmaxf(rintf(lA[(size_t)k * 16 + r] * inv), -QMAXF), QMAXF);
  aqT[e] = (signed char)(int)q;
}

__global__ __launch_bounds__(256) void k_quant_lB(const float* __restrict__ lB,
                                                  signed char* __restrict__ bqT,
                                                  const float* __restrict__ scales) {
  const float inv = scales[3];
  const int e = blockIdx.x * 256 + threadIdx.x;
  const int n = e >> 4, r = e & 15;
  float q = fminf(fmaxf(rintf(lB[(size_t)r * D_OUT + n] * inv), -QMAXF), QMAXF);
  bqT[e] = (signed char)(int)q;
}

__global__ __launch_bounds__(256) void k_lora_t(const signed char* __restrict__ xq,
                                                const signed char* __restrict__ aqT,
                                                int* __restrict__ t) {
  const int tid = threadIdx.x, wave = tid >> 5, lane = tid & 31;
  const int m0 = blockIdx.x * 128 + wave * 16;
  v8i acc = {0, 0, 0, 0, 0, 0, 0, 0};
  for (int k0 = 0; k0 < D_IN; k0 += 64) {
    v8i a = load_a_frag(xq, m0, k0, lane);
    v8i b = load_b_frag(aqT, 0, k0, lane, D_IN);
    acc = __builtin_amdgcn_wmma_i32_16x16x64_iu8(true, a, true, b, acc, false, false);
  }
  const int r = lane & 15, hi = lane >> 4;
#pragma unroll
  for (int v = 0; v < 8; ++v) {
    int m = m0 + hi * 8 + v;
    t[(size_t)m * 16 + r] = acc[v];
  }
}

__global__ __launch_bounds__(256) void k_gemm(const signed char* __restrict__ xq,
                                              const signed char* __restrict__ wq,
                                              const float* __restrict__ xscale,
                                              const float* __restrict__ bias,
                                              const float* __restrict__ scales,
                                              const int* __restrict__ t,
                                              const signed char* __restrict__ bqT,
                                              float* __restrict__ out) {
  __shared__ __align__(16) signed char sA[2][128 * LDS_STRIDE];
  __shared__ __align__(16) signed char sB[2][128 * LDS_STRIDE];

  const int tid = threadIdx.x, wave = tid >> 5, lane = tid & 31;
  const int bm = blockIdx.x * 128, bn = blockIdx.y * 128;
  const int wmL = (wave >> 1) * 32;
  const int wnL = (wave & 1) * 64;
  const int cr = tid >> 2;
  const int cc = (tid & 3) * 16;

  v8i acc[2][4];
#pragma unroll
  for (int i = 0; i < 2; ++i)
#pragma unroll
    for (int j = 0; j < 4; ++j) {
      v8i z = {0, 0, 0, 0, 0, 0, 0, 0};
      acc[i][j] = z;
    }

#if HAVE_ASYNC_LDS
#pragma unroll
  for (int u = 0; u < 2; ++u) {
    int r = cr + 64 * u;
    async_copy16(xq + (size_t)(bm + r) * D_IN + cc, &sA[0][r * LDS_STRIDE + cc]);
    async_copy16(wq + (size_t)(bn + r) * D_IN + cc, &sB[0][r * LDS_STRIDE + cc]);
  }
  wait_async0();
#else
  int4 ra[2], rb[2];
#pragma unroll
  for (int u = 0; u < 2; ++u) {
    int r = cr + 64 * u;
    ra[u] = *(const int4*)(xq + (size_t)(bm + r) * D_IN + cc);
    rb[u] = *(const int4*)(wq + (size_t)(bn + r) * D_IN + cc);
  }
#pragma unroll
  for (int u = 0; u < 2; ++u) {
    int r = cr + 64 * u;
    *(int4*)(&sA[0][r * LDS_STRIDE + cc]) = ra[u];
    *(int4*)(&sB[0][r * LDS_STRIDE + cc]) = rb[u];
  }
#endif
  __syncthreads();

  int buf = 0;
  for (int k0 = 0; k0 < D_IN; k0 += 64) {
    const bool next = (k0 + 64) < D_IN;
#if HAVE_ASYNC_LDS
    if (next) {
      const int kk = k0 + 64, nb = buf ^ 1;
#pragma unroll
      for (int u = 0; u < 2; ++u) {
        int r = cr + 64 * u;
        async_copy16(xq + (size_t)(bm + r) * D_IN + kk + cc, &sA[nb][r * LDS_STRIDE + cc]);
        async_copy16(wq + (size_t)(bn + r) * D_IN + kk + cc, &sB[nb][r * LDS_STRIDE + cc]);
      }
    }
#else
    if (next) {
      const int kk = k0 + 64;
#pragma unroll
      for (int u = 0; u < 2; ++u) {
        int r = cr + 64 * u;
        ra[u] = *(const int4*)(xq + (size_t)(bm + r) * D_IN + kk + cc);
        rb[u] = *(const int4*)(wq + (size_t)(bn + r) * D_IN + kk + cc);
      }
    }
#endif
    {
      v8i a0 = lds_a_frag(sA[buf], wmL,      lane);
      v8i a1 = lds_a_frag(sA[buf], wmL + 16, lane);
      v8i b0 = lds_b_frag(sB[buf], wnL,      lane);
      v8i b1 = lds_b_frag(sB[buf], wnL + 16, lane);
      v8i b2 = lds_b_frag(sB[buf], wnL + 32, lane);
      v8i b3 = lds_b_frag(sB[buf], wnL + 48, lane);
      acc[0][0] = __builtin_amdgcn_wmma_i32_16x16x64_iu8(true, a0, true, b0, acc[0][0], false, false);
      acc[0][1] = __builtin_amdgcn_wmma_i32_16x16x64_iu8(true, a0, true, b1, acc[0][1], false, false);
      acc[0][2] = __builtin_amdgcn_wmma_i32_16x16x64_iu8(true, a0, true, b2, acc[0][2], false, false);
      acc[0][3] = __builtin_amdgcn_wmma_i32_16x16x64_iu8(true, a0, true, b3, acc[0][3], false, false);
      acc[1][0] = __builtin_amdgcn_wmma_i32_16x16x64_iu8(true, a1, true, b0, acc[1][0], false, false);
      acc[1][1] = __builtin_amdgcn_wmma_i32_16x16x64_iu8(true, a1, true, b1, acc[1][1], false, false);
      acc[1][2] = __builtin_amdgcn_wmma_i32_16x16x64_iu8(true, a1, true, b2, acc[1][2], false, false);
      acc[1][3] = __builtin_amdgcn_wmma_i32_16x16x64_iu8(true, a1, true, b3, acc[1][3], false, false);
    }
#if HAVE_ASYNC_LDS
    if (next) wait_async0();
    __syncthreads();
#else
    if (next) {
      const int nb = buf ^ 1;
#pragma unroll
      for (int u = 0; u < 2; ++u) {
        int r = cr + 64 * u;
        *(int4*)(&sA[nb][r * LDS_STRIDE + cc]) = ra[u];
        *(int4*)(&sB[nb][r * LDS_STRIDE + cc]) = rb[u];
      }
    }
    __syncthreads();
#endif
    buf ^= 1;
  }

  const float w_s = scales[0];
  const float lf  = scales[4];
  const int hi = lane >> 4, nc = lane & 15;

  float bqf[4][16];
  float bsv[4];
#pragma unroll
  for (int j = 0; j < 4; ++j) {
    const int n = bn + wnL + 16 * j + nc;
    int4 bw = *(const int4*)(bqT + (size_t)n * 16);
    int wds[4] = {bw.x, bw.y, bw.z, bw.w};
#pragma unroll
    for (int bb = 0; bb < 16; ++bb)
      bqf[j][bb] = (float)((wds[bb >> 2] << (24 - 8 * (bb & 3))) >> 24);
    bsv[j] = bias[n];
  }

#pragma unroll
  for (int i = 0; i < 2; ++i) {
#pragma unroll
    for (int v = 0; v < 8; ++v) {
      const int m = bm + wmL + 16 * i + hi * 8 + v;
      const int4* tp = (const int4*)(t + (size_t)m * 16);
      int4 t0 = tp[0], t1 = tp[1], t2 = tp[2], t3 = tp[3];
      float tf[16] = {(float)t0.x, (float)t0.y, (float)t0.z, (float)t0.w,
                      (float)t1.x, (float)t1.y, (float)t1.z, (float)t1.w,
                      (float)t2.x, (float)t2.y, (float)t2.z, (float)t2.w,
                      (float)t3.x, (float)t3.y, (float)t3.z, (float)t3.w};
      const float xs = xscale[m];
      const float bscale = w_s * xs, lscale = lf * xs;
      float* orow = out + (size_t)m * D_OUT + bn + wnL + nc;
#pragma unroll
      for (int j = 0; j < 4; ++j) {
        float ls = 0.f;
#pragma unroll
        for (int r = 0; r < 16; ++r) ls = fmaf(tf[r], bqf[j][r], ls);
        orow[16 * j] = (float)acc[i][j][v] * bscale + ls * lscale + bsv[j];
      }
    }
  }
}

extern "C" void kernel_launch(void* const* d_in, const int* in_sizes, int n_in,
                              void* d_out, int out_size, void* d_ws, size_t ws_size,
                              hipStream_t stream) {
  (void)in_sizes; (void)n_in; (void)out_size; (void)ws_size;
  const float* x    = (const float*)d_in[0];
  const float* w    = (const float*)d_in[1];
  const float* bias = (const float*)d_in[2];
  const float* lA   = (const float*)d_in[3];
  const float* lB   = (const float*)d_in[4];
  float* out = (float*)d_out;

  char* ws = (char*)d_ws;
  float*       scales = (float*)(ws + OFF_SCALES);
  float*       part   = (float*)(ws + OFF_PART);
  float*       xscale = (float*)(ws + OFF_XSCALE);
  int*         tbuf   = (int*)  (ws + OFF_T);
  signed char* aqT    = (signed char*)(ws + OFF_AQT);
  signed char* bqT    = (signed char*)(ws + OFF_BQT);
  signed char* xq     = (signed char*)(ws + OFF_XQ);
  signed char* wq     = (signed char*)(ws + OFF_WQ);

  k_quant_act<<<M_TOK, 256, 0, stream>>>(x, xq, xscale);
  k_wabs<<<1024, 256, 0, stream>>>(w, part);
  k_finalize<<<1, 256, 0, stream>>>(part, lA, lB, scales);
  k_quant_w<<<16384, 256, 0, stream>>>(w, wq, scales);
  k_quant_lA<<<256, 256, 0, stream>>>(lA, aqT, scales);
  k_quant_lB<<<256, 256, 0, stream>>>(lB, bqT, scales);
  k_lora_t<<<M_TOK / 128, 256, 0, stream>>>(xq, aqT, tbuf);
  k_gemm<<<dim3(M_TOK / 128, D_OUT / 128), 256, 0, stream>>>(xq, wq, xscale, bias, scales,
                                                             tbuf, bqT, out);
}
